// BilevelGraphAttention_37134287242023
// MI455X (gfx1250) — matrix operand order repaired and checked by reading, not compiled, not run on hardware
//
#include <hip/hip_runtime.h>
#include <math.h>

typedef __attribute__((ext_vector_type(16))) __bf16 v16bf;
typedef __attribute__((ext_vector_type(8)))  float  v8f;

#define DEV __device__ __forceinline__

// ---------------- problem sizes ----------------
constexpr int N_RES       = 512;
constexpr int N_RES_EDGES = 2048;
constexpr int N_ATOMS     = 7168;          // 512 * 14
constexpr int E_ATOM      = 401408;        // 2048 * 14 * 14

// ---------------- workspace layout (in floats) ----------------
constexpr size_t OFF_Q    = 0;                                  // 7168*128
constexpr size_t OFF_KV   = OFF_Q    + (size_t)N_ATOMS * 128;   // 7168*256
constexpr size_t OFF_R    = OFF_KV   + (size_t)N_ATOMS * 256;   // E*4 (R, then exp(R-max))
constexpr size_t OFF_SMAX = OFF_R    + (size_t)E_ATOM  * 4;     // 7168*4
constexpr size_t OFF_SSUM = OFF_SMAX + (size_t)N_ATOMS * 4;     // 7168*4
constexpr size_t OFF_BSUM = OFF_SSUM + (size_t)N_ATOMS * 4;     // 2048*4
constexpr size_t OFF_CNT  = OFF_BSUM + (size_t)N_RES_EDGES * 4; // 2048
constexpr size_t OFF_RB   = OFF_CNT  + (size_t)N_RES_EDGES;     // 2048*4
constexpr size_t OFF_BETA = OFF_RB   + (size_t)N_RES_EDGES * 4; // 2048*4
constexpr size_t OFF_UPD  = OFF_BETA + (size_t)N_RES_EDGES * 4; // 7168*128

// ---------------- helpers ----------------
DEV v8f wmma_bf16(v16bf a, v16bf b, v8f c) {
    // D = A(16x32 bf16) * B(32x16 bf16) + C(16x16 f32)
    return __builtin_amdgcn_wmma_f32_16x16x32_bf16(
        false, a, false, b, (short)0, c, false, false);
}

DEV void atomAddF(float* p, float v) {
    (void)__hip_atomic_fetch_add(p, v, __ATOMIC_RELAXED, __HIP_MEMORY_SCOPE_AGENT);
}

DEV void atomMaxF(float* p, float v) {
    unsigned* u = (unsigned*)p;
    unsigned old = __hip_atomic_load(u, __ATOMIC_RELAXED, __HIP_MEMORY_SCOPE_AGENT);
    while (__uint_as_float(old) < v) {
        unsigned assumed = old;
        old = atomicCAS(u, assumed, __float_as_uint(v));
        if (old == assumed) break;
    }
}

// Pack 4 contiguous f32 (as float4) into bf16 vector slots [j..j+3]
DEV void put4(v16bf& dst, int j, float4 t) {
    dst[j + 0] = (__bf16)t.x; dst[j + 1] = (__bf16)t.y;
    dst[j + 2] = (__bf16)t.z; dst[j + 3] = (__bf16)t.w;
}

// Load A fragment (16x32 bf16) for a row-major f32 matrix row `ar` at K-chunk base kb0.
// Layout: a[0..7] = ar[kb0 + half*8 .. +7]; a[8..15] = ar[kb0 + 16 + half*8 .. +7]
DEV v16bf load_a_frag(const float* ar, int kb0, int half) {
    const float4* p0 = (const float4*)(ar + kb0 + (half << 3));
    const float4* p1 = (const float4*)(ar + kb0 + 16 + (half << 3));
    v16bf a;
    put4(a, 0, p0[0]); put4(a, 4, p0[1]);
    put4(a, 8, p1[0]); put4(a, 12, p1[1]);
    return a;
}

// Load B fragment: b[0..7] = wr[kb0 + half*8 .. +7]; b[8..15] = wr[kb0 + 16 + half*8 .. +7]
DEV v16bf load_b_frag(const float* wr, int kb0, int half) {
    const float4* p = (const float4*)(wr + kb0 + (half << 3));
    const float4* q = (const float4*)(wr + kb0 + 16 + (half << 3));
    v16bf b;
    put4(b, 0, p[0]); put4(b, 4, p[1]);
    put4(b, 8, q[0]); put4(b, 12, q[1]);
    return b;
}

// ---------------- generic fill ----------------
__global__ void fill_kernel(float* __restrict__ p, float v, int n) {
    int i = blockIdx.x * blockDim.x + threadIdx.x;
    if (i < n) p[i] = v;
}

// ---------------- K0: q/kv projection (WMMA) ----------------
// q  = atom_features(7168x32) @ W_q.T(32x128)  + b_q
// kv = atom_features(7168x32) @ W_kv.T(32x256) + b_kv
// Combined N = 384 = 24 tiles of 16; M = 448 tiles of 16; K = 32 -> one WMMA/wave.
__global__ __launch_bounds__(128)
void qkv_proj_kernel(const float* __restrict__ atomf,
                     const float* __restrict__ Wq,  const float* __restrict__ bq,
                     const float* __restrict__ Wkv, const float* __restrict__ bkv,
                     float* __restrict__ qbuf, float* __restrict__ kvbuf) {
    const int lane = threadIdx.x & 31;
    const int wv   = threadIdx.x >> 5;               // 4 waves per block
    const int m0   = blockIdx.x * 16;
    const int n0   = (blockIdx.y * 4 + wv) * 16;
    const int half = lane >> 4;
    const int nlo  = lane & 15;

    // A fragment
    const float* ar = atomf + (size_t)(m0 + nlo) * 32;
    const v16bf a = load_a_frag(ar, 0, half);

    // Branch-free source/dest select (q region: n < 128)
    const int  n    = n0 + nlo;
    const bool is_q = (n < 128);
    const float* wr    = is_q ? (Wq + (size_t)n * 32) : (Wkv + (size_t)(n - 128) * 32);
    const float* bptr  = is_q ? (bq + n) : (bkv + (n - 128));
    float*       obase = is_q ? (qbuf + n) : (kvbuf + (n - 128));
    const int    ostr  = is_q ? 128 : 256;

    const v16bf b = load_b_frag(wr, 0, half);

    v8f c = {};
    c = wmma_bf16(a, b, c);

    const float bias = *bptr;
    float* p = obase + (size_t)(m0 + (half << 3)) * ostr;
#pragma unroll
    for (int i = 0; i < 8; ++i) {
        *p = c[i] + bias;
        p += ostr;
    }
}

// ---------------- K1: res-level bias (WMMA + LDS second layer) ----------------
// cat(2048x320) = [res_edge_features | node[idx0] | node[idx1]]
// h = relu(cat @ W_rb0.T(320x32) + b_rb0);  resbias = h @ W_rb1.T(32x4) + b_rb1
__global__ __launch_bounds__(32)
void res_bias_kernel(const float* __restrict__ nodef,
                     const float* __restrict__ refe,
                     const int*   __restrict__ rei,
                     const float* __restrict__ W0, const float* __restrict__ b0,
                     const float* __restrict__ W1, const float* __restrict__ b1,
                     float* __restrict__ resbias) {
    __shared__ float sh[16 * 32];
    const int lane = threadIdx.x & 31;
    const int half = lane >> 4;
    const int nlo  = lane & 15;
    const int m0   = blockIdx.x * 16;
    const int re   = m0 + nlo;
    const int i0   = rei[re];
    const int i1   = rei[N_RES_EDGES + re];

    const float* row_ref = refe  + (size_t)re * 64;
    const float* row_n0  = nodef + (size_t)i0 * 128;
    const float* row_n1  = nodef + (size_t)i1 * 128;

    v8f c0 = {}, c1 = {};
#pragma unroll
    for (int kk = 0; kk < 10; ++kk) {            // K = 320 = 10 * 32
        const int fb = kk * 32;
        // Each 32-wide chunk lies fully inside one concat region
        // (boundaries at 64 and 192 are multiples of 32) -> folds at compile time.
        const float* base;
        int off;
        if (fb < 64)       { base = row_ref; off = fb; }
        else if (fb < 192) { base = row_n0;  off = fb - 64; }
        else               { base = row_n1;  off = fb - 192; }
        const v16bf a = load_a_frag(base, off, half);

#pragma unroll
        for (int t = 0; t < 2; ++t) {            // N = 32 = 2 tiles of 16
            const int nn = t * 16 + nlo;
            const float* wr = W0 + (size_t)nn * 320 + fb;
            const v16bf b = load_b_frag(wr, 0, half);
            if (t == 0) c0 = wmma_bf16(a, b, c0);
            else        c1 = wmma_bf16(a, b, c1);
        }
    }

    const float bb0 = b0[nlo], bb1 = b0[16 + nlo];
#pragma unroll
    for (int i = 0; i < 8; ++i) {
        const int r = (half << 3) + i;
        sh[r * 32 + nlo]      = fmaxf(c0[i] + bb0, 0.f);
        sh[r * 32 + 16 + nlo] = fmaxf(c1[i] + bb1, 0.f);
    }
    __syncthreads();

    // second layer: 16 rows x 4 outs = 64 dots of length 32, 2 per lane
    const int r = lane >> 1;
#pragma unroll
    for (int oo = 0; oo < 2; ++oo) {
        const int o = ((lane & 1) << 1) + oo;
        float s = b1[o];
#pragma unroll
        for (int j = 0; j < 32; ++j) s += sh[r * 32 + j] * W1[o * 32 + j];
        resbias[(size_t)(m0 + r) * 4 + o] = s;
    }
}

// ---------------- K2a: per-edge R = qk/sqrt(32) + res_bias + dist_bias ----------------
__global__ __launch_bounds__(256)
void edge_R_kernel(const float* __restrict__ coords,
                   const int*   __restrict__ cae,
                   const int*   __restrict__ aere,
                   const float* __restrict__ Wde0, const float* __restrict__ bde0,
                   const float* __restrict__ Wde1, const float* __restrict__ bde1,
                   const float* __restrict__ Wde2, const float* __restrict__ bde2,
                   const float* __restrict__ qbuf, const float* __restrict__ kvbuf,
                   const float* __restrict__ resbias,
                   float* __restrict__ Rbuf,  float* __restrict__ segmax,
                   float* __restrict__ bsum,  float* __restrict__ cnt) {
    __shared__ float sW0[256], sW1[256], sW2[64], sb0[16], sb1[16], sb2[4];
    const int t = threadIdx.x;
    if (t < 256) { sW0[t] = Wde0[t]; sW1[t] = Wde1[t]; }
    if (t < 64)  sW2[t] = Wde2[t];
    if (t < 16)  { sb0[t] = bde0[t]; sb1[t] = bde1[t]; }
    if (t < 4)   sb2[t] = bde2[t];
    __syncthreads();

    const int e    = blockIdx.x * 256 + t;
    const int dstA = cae[e];              // row 0 = dst
    const int srcA = cae[E_ATOM + e];     // row 1 = src
    const int re   = aere[e];

    const float vx = coords[(size_t)dstA * 3 + 0] - coords[(size_t)srcA * 3 + 0] + 1e-8f;
    const float vy = coords[(size_t)dstA * 3 + 1] - coords[(size_t)srcA * 3 + 1] + 1e-8f;
    const float vz = coords[(size_t)dstA * 3 + 2] - coords[(size_t)srcA * 3 + 2] + 1e-8f;
    const float d = sqrtf(vx * vx + vy * vy + vz * vz);

    float r[16];
#pragma unroll
    for (int j = 0; j < 16; ++j) {
        const float mu = (20.f / 15.f) * (float)j;   // linspace(0,20,16)
        const float z  = (d - mu) * 0.8f;            // / sigma, sigma = 1.25
        r[j] = __expf(-z * z);
    }
    float h1[16];
#pragma unroll
    for (int o = 0; o < 16; ++o) {
        float s = sb0[o];
#pragma unroll
        for (int j = 0; j < 16; ++j) s += sW0[o * 16 + j] * r[j];
        h1[o] = fmaxf(s, 0.f);
    }
    float h2[16];
#pragma unroll
    for (int o = 0; o < 16; ++o) {
        float s = sb1[o];
#pragma unroll
        for (int j = 0; j < 16; ++j) s += sW1[o * 16 + j] * h1[j];
        h2[o] = fmaxf(s, 0.f);
    }

    const float4* qp = (const float4*)(qbuf  + (size_t)srcA * 128);
    const float4* kp = (const float4*)(kvbuf + (size_t)dstA * 256);
#pragma unroll
    for (int h = 0; h < 4; ++h) {
        float s = 0.f;
#pragma unroll
        for (int u = 0; u < 8; ++u) {
            const float4 q4 = qp[h * 8 + u];
            const float4 k4 = kp[h * 16 + u];   // k = kv[..., :32]
            s += q4.x * k4.x + q4.y * k4.y + q4.z * k4.z + q4.w * k4.w;
        }
        float db = sb2[h];
#pragma unroll
        for (int j = 0; j < 16; ++j) db += sW2[h * 16 + j] * h2[j];
        const float R = s * 0.17677669529663687f + resbias[(size_t)re * 4 + h] + db;
        Rbuf[(size_t)e * 4 + h] = R;
        atomMaxF(&segmax[(size_t)srcA * 4 + h], R);
        atomAddF(&bsum[(size_t)re * 4 + h], R);
    }
    atomAddF(&cnt[re], 1.f);
}

// ---------------- K2b: e = exp(R - max), segment sums over src ----------------
__global__ __launch_bounds__(256)
void edge_exp_kernel(const int* __restrict__ cae,
                     const float* __restrict__ segmax,
                     float* __restrict__ Rbuf, float* __restrict__ segsum) {
    const int idx = blockIdx.x * 256 + threadIdx.x;   // over E*4
    const int e = idx >> 2, h = idx & 3;
    const int srcA = cae[E_ATOM + e];
    const float m  = segmax[(size_t)srcA * 4 + h];
    const float ex = __expf(Rbuf[idx] - m);
    Rbuf[idx] = ex;
    atomAddF(&segsum[(size_t)srcA * 4 + h], ex);
}

// ---------------- K2c: beta = segment-softmax(block_r, res_edge_index[1]) ----------------
__global__ __launch_bounds__(256)
void beta_kernel(const int* __restrict__ rei,
                 const float* __restrict__ bsum, const float* __restrict__ cnt,
                 float* __restrict__ beta) {
    const int r = blockIdx.x * blockDim.x + threadIdx.x;
    if (r >= N_RES) return;
    float mx[4] = {-1e30f, -1e30f, -1e30f, -1e30f};
    for (int re = 0; re < N_RES_EDGES; ++re)
        if (rei[N_RES_EDGES + re] == r) {
            const float c = fmaxf(cnt[re], 1.f);
            for (int h = 0; h < 4; ++h) mx[h] = fmaxf(mx[h], bsum[re * 4 + h] / c);
        }
    float sm[4] = {0.f, 0.f, 0.f, 0.f};
    for (int re = 0; re < N_RES_EDGES; ++re)
        if (rei[N_RES_EDGES + re] == r) {
            const float c = fmaxf(cnt[re], 1.f);
            for (int h = 0; h < 4; ++h) sm[h] += __expf(bsum[re * 4 + h] / c - mx[h]);
        }
    for (int re = 0; re < N_RES_EDGES; ++re)
        if (rei[N_RES_EDGES + re] == r) {
            const float c = fmaxf(cnt[re], 1.f);
            for (int h = 0; h < 4; ++h)
                beta[re * 4 + h] = __expf(bsum[re * 4 + h] / c - mx[h]) / (sm[h] + 1e-16f);
        }
}

// ---------------- K2d: upd[src] += alpha * beta * v ----------------
__global__ __launch_bounds__(256)
void edge_upd_kernel(const int* __restrict__ cae, const int* __restrict__ aere,
                     const float* __restrict__ Rbuf, const float* __restrict__ segsum,
                     const float* __restrict__ beta, const float* __restrict__ kvbuf,
                     float* __restrict__ upd) {
    const int idx = blockIdx.x * 256 + threadIdx.x;   // over E*4
    const int e = idx >> 2, h = idx & 3;
    const int srcA = cae[E_ATOM + e];
    const int dstA = cae[e];
    const int re   = aere[e];
    const float alpha = Rbuf[idx] / (segsum[(size_t)srcA * 4 + h] + 1e-16f);
    const float w     = alpha * beta[(size_t)re * 4 + h];
    const float4* vp = (const float4*)(kvbuf + (size_t)dstA * 256 + h * 64 + 32); // v half
    float* up = upd + (size_t)srcA * 128 + h * 32;
#pragma unroll
    for (int u = 0; u < 8; ++u) {
        const float4 v4 = vp[u];
        atomAddF(&up[u * 4 + 0], w * v4.x);
        atomAddF(&up[u * 4 + 1], w * v4.y);
        atomAddF(&up[u * 4 + 2], w * v4.z);
        atomAddF(&up[u * 4 + 3], w * v4.w);
    }
}

// ---------------- K3: output projection (WMMA) ----------------
// out(7168x32) = upd(7168x128) @ W_o.T(128x32) + b_o
__global__ __launch_bounds__(64)
void out_proj_kernel(const float* __restrict__ upd, const float* __restrict__ Wo,
                     const float* __restrict__ bo, float* __restrict__ out) {
    const int lane = threadIdx.x & 31;
    const int wv   = threadIdx.x >> 5;    // 2 waves per block: n-tiles 0/1
    const int m0   = blockIdx.x * 16;
    const int half = lane >> 4;
    const int nlo  = lane & 15;
    const int n    = wv * 16 + nlo;
    const float* ar = upd + (size_t)(m0 + nlo) * 128;
    const float* wr = Wo + (size_t)n * 128;

    v8f c = {};
#pragma unroll
    for (int kk = 0; kk < 4; ++kk) {       // K = 128 = 4 * 32
        const int fb = kk * 32;
        const v16bf a = load_a_frag(ar, fb, half);
        const v16bf b = load_b_frag(wr, fb, half);
        c = wmma_bf16(a, b, c);
    }
    const float bias = bo[n];
#pragma unroll
    for (int i = 0; i < 8; ++i)
        out[(size_t)(m0 + (half << 3) + i) * 32 + n] = c[i] + bias;
}

// ---------------- launch ----------------
extern "C" void kernel_launch(void* const* d_in, const int* in_sizes, int n_in,
                              void* d_out, int out_size, void* d_ws, size_t ws_size,
                              hipStream_t stream) {
    (void)in_sizes; (void)n_in; (void)out_size; (void)ws_size;

    const float* nodef  = (const float*)d_in[0];
    const float* refe   = (const float*)d_in[1];
    const float* atomf  = (const float*)d_in[2];
    const float* coords = (const float*)d_in[3];
    const int*   rei    = (const int*)d_in[4];
    const int*   cae    = (const int*)d_in[5];
    const int*   aere   = (const int*)d_in[6];
    const float* Wde0 = (const float*)d_in[7];  const float* bde0 = (const float*)d_in[8];
    const float* Wde1 = (const float*)d_in[9];  const float* bde1 = (const float*)d_in[10];
    const float* Wde2 = (const float*)d_in[11]; const float* bde2 = (const float*)d_in[12];
    const float* Wrb0 = (const float*)d_in[13]; const float* brb0 = (const float*)d_in[14];
    const float* Wrb1 = (const float*)d_in[15]; const float* brb1 = (const float*)d_in[16];
    const float* Wq   = (const float*)d_in[17]; const float* bq   = (const float*)d_in[18];
    const float* Wkv  = (const float*)d_in[19]; const float* bkv  = (const float*)d_in[20];
    const float* Wo   = (const float*)d_in[21]; const float* bo   = (const float*)d_in[22];

    float* ws     = (float*)d_ws;
    float* qbuf   = ws + OFF_Q;
    float* kvbuf  = ws + OFF_KV;
    float* Rbuf   = ws + OFF_R;
    float* segmax = ws + OFF_SMAX;
    float* segsum = ws + OFF_SSUM;
    float* bsum   = ws + OFF_BSUM;
    float* cnt    = ws + OFF_CNT;
    float* resb   = ws + OFF_RB;
    float* beta   = ws + OFF_BETA;
    float* upd    = ws + OFF_UPD;
    float* outp   = (float*)d_out;

    auto fill = [&](float* p, float v, int n) {
        fill_kernel<<<(n + 255) / 256, 256, 0, stream>>>(p, v, n);
    };
    fill(segmax, -1e30f, N_ATOMS * 4);
    fill(segsum, 0.f,    N_ATOMS * 4);
    fill(bsum,   0.f,    N_RES_EDGES * 4);
    fill(cnt,    0.f,    N_RES_EDGES);
    fill(upd,    0.f,    N_ATOMS * 128);

    qkv_proj_kernel<<<dim3(N_ATOMS / 16, 6), 128, 0, stream>>>(
        atomf, Wq, bq, Wkv, bkv, qbuf, kvbuf);
    res_bias_kernel<<<N_RES_EDGES / 16, 32, 0, stream>>>(
        nodef, refe, rei, Wrb0, brb0, Wrb1, brb1, resb);
    edge_R_kernel<<<E_ATOM / 256, 256, 0, stream>>>(
        coords, cae, aere, Wde0, bde0, Wde1, bde1, Wde2, bde2,
        qbuf, kvbuf, resb, Rbuf, segmax, bsum, cnt);
    edge_exp_kernel<<<(E_ATOM * 4) / 256, 256, 0, stream>>>(cae, segmax, Rbuf, segsum);
    beta_kernel<<<2, 256, 0, stream>>>(rei, bsum, cnt, beta);
    edge_upd_kernel<<<(E_ATOM * 4) / 256, 256, 0, stream>>>(
        cae, aere, Rbuf, segsum, beta, kvbuf, upd);
    out_proj_kernel<<<N_ATOMS / 16, 64, 0, stream>>>(upd, Wo, bo, outp);
}
